// BiasedMHSA_60627758350797
// MI455X (gfx1250) — compile-verified, matrix operand order checked by reading, not run on hardware
//
#include <hip/hip_runtime.h>

#define D_MODEL 1024
#define NHEAD   16
#define DK      64
#define BATCH   2
#define SEQ     2048
#define NROWS   (BATCH * SEQ)   // 4096

typedef __attribute__((ext_vector_type(16))) __bf16 v16bf;
typedef __attribute__((ext_vector_type(8)))  __bf16 v8bf;
typedef __attribute__((ext_vector_type(8)))  float  v8f;

typedef int v4i_t __attribute__((vector_size(16)));
typedef __attribute__((address_space(1))) v4i_t* gv4p;   // global int4*
typedef __attribute__((address_space(3))) v4i_t* lv4p;   // LDS int4*

union FragU  { v16bf v; v8bf h[2]; };
union BfBits { unsigned short u; __bf16 b; };

// fp32 -> bf16, round-to-nearest-even
__device__ __forceinline__ __bf16 f2bf(float f) {
    unsigned u = __float_as_uint(f);
    u = u + 0x7FFFu + ((u >> 16) & 1u);
    BfBits r; r.u = (unsigned short)(u >> 16);
    return r.b;
}

// ---- CDNA5 async global->LDS copy (16 bytes per lane), ASYNCcnt-tracked ----
__device__ __forceinline__ void async_copy16(const __bf16* g, const __bf16* l) {
    unsigned lds_lo = (unsigned)(uintptr_t)l;  // generic LDS ptr: low 32 = ds addr
#if __has_builtin(__builtin_amdgcn_global_load_async_to_lds_b128)
    __builtin_amdgcn_global_load_async_to_lds_b128(
        (gv4p)(uintptr_t)g, (lv4p)(uintptr_t)lds_lo, 0, 0);
#else
    unsigned long long ga = (unsigned long long)(uintptr_t)g;
    asm volatile("global_load_async_to_lds_b128 %0, %1, off"
                 :: "v"(lds_lo), "v"(ga) : "memory");
#endif
}

#if __has_builtin(__builtin_amdgcn_s_wait_asynccnt)
#define WAIT_ASYNC(n) __builtin_amdgcn_s_wait_asynccnt(n)
#else
#define WAIT_ASYNC(n) asm volatile("s_wait_asynccnt %0" :: "i"(n) : "memory")
#endif

// Load a 16x32 bf16 tile (row-major, `stride` elements per row) into the
// CDNA5 WMMA 16-bit A-layout (serves B when source is B-transposed row-major):
//   lane L: row m = L&15 ; K = (L>=16 ? 8 : 0) + {0..7, 16..23}
__device__ __forceinline__ v16bf load_frag(const __bf16* base, int stride) {
    const int lane = threadIdx.x & 31;
    const int m  = lane & 15;
    const int hi = lane >> 4;
    const __bf16* r = base + (size_t)m * stride + hi * 8;
    FragU f;
    f.h[0] = *reinterpret_cast<const v8bf*>(r);
    f.h[1] = *reinterpret_cast<const v8bf*>(r + 16);
    return f.v;
}

__device__ __forceinline__ v8f wmma_bf16(v16bf a, v16bf b, v8f c) {
    return __builtin_amdgcn_wmma_f32_16x16x32_bf16(
        false, a, false, b, (short)0, c, false, false);
}

// ---------------------------------------------------------------- converts
__global__ void cvt_f32_to_bf16(const float* __restrict__ src,
                                __bf16* __restrict__ dst, int n) {
    for (int i = blockIdx.x * blockDim.x + threadIdx.x; i < n;
         i += gridDim.x * blockDim.x)
        dst[i] = f2bf(src[i]);
}

// ------------------------------------------------------- fused QKV GEMM
// y = x @ W^T + b ; block = 8 waves = 128(M) x 64(N), LDS double-buffered
// async staging. blockIdx.z: 0=Q (pre-scaled 1/sqrt(dk)), 1=K, 2=V (transposed).
__global__ __launch_bounds__(256) void qkv_gemm(
    const __bf16* __restrict__ xbf,
    const __bf16* __restrict__ Wq, const __bf16* __restrict__ Wk,
    const __bf16* __restrict__ Wv,
    const float* __restrict__ bq, const float* __restrict__ bk,
    const float* __restrict__ bv,
    __bf16* __restrict__ Qo, __bf16* __restrict__ Ko, __bf16* __restrict__ Vt)
{
    __shared__ __attribute__((aligned(16))) __bf16 As[2][128][32]; // 16 KB
    __shared__ __attribute__((aligned(16))) __bf16 Bs[2][64][32];  //  8 KB

    const int tid  = threadIdx.x;
    const int w    = tid >> 5;
    const int lane = tid & 31;
    const int col  = lane & 15;
    const int hi   = lane >> 4;
    const int rowB = blockIdx.x * 128;           // block row base
    const int n0   = blockIdx.y * 64;
    const int z    = blockIdx.z;

    const __bf16* W    = (z == 0) ? Wq : (z == 1) ? Wk : Wv;
    const float*  bias = (z == 0) ? bq : (z == 1) ? bk : bv;

    // 3 async b128 per thread per chunk: 2 for the 128x32 A tile, 1 for 64x32 B
    auto stage = [&](int kc, int buf) {
#pragma unroll
        for (int s = 0; s < 2; ++s) {
            const int seg = tid * 2 + s;
            const int r = seg >> 2, c = (seg & 3) * 8;
            async_copy16(xbf + (size_t)(rowB + r) * D_MODEL + kc + c,
                         &As[buf][r][c]);
        }
        const int r = tid >> 2, c = (tid & 3) * 8;
        async_copy16(W + (size_t)(n0 + r) * D_MODEL + kc + c, &Bs[buf][r][c]);
    };

    v8f acc[4] = {};
    stage(0, 0);
    for (int it = 0; it < D_MODEL / 32; ++it) {
        const int buf = it & 1;
        if (it + 1 < D_MODEL / 32) { stage((it + 1) * 32, buf ^ 1); WAIT_ASYNC(3); }
        else                       { WAIT_ASYNC(0); }
        __syncthreads();

        v16bf a = load_frag(&As[buf][w * 16][0], 32);
#pragma unroll
        for (int n = 0; n < 4; ++n) {
            v16bf bb = load_frag(&Bs[buf][n * 16][0], 32);
            acc[n] = wmma_bf16(a, bb, acc[n]);
        }
        __syncthreads();
    }

    const int row0 = rowB + w * 16;
#pragma unroll
    for (int n = 0; n < 4; ++n) {
        const int   c    = n0 + n * 16 + col;
        const float bc   = bias[c];
        const int   h    = c >> 6;
        const int   dkid = c & 63;
#pragma unroll
        for (int j = 0; j < 8; ++j) {
            const int row = row0 + j + hi * 8;
            const int b   = row >> 11;     // /SEQ
            const int t   = row & 2047;    // %SEQ
            const int bh  = b * NHEAD + h;
            const float v = acc[n][j] + bc;
            if (z == 0)
                Qo[((size_t)bh * SEQ + t) * DK + dkid] = f2bf(v * 0.125f);
            else if (z == 1)
                Ko[((size_t)bh * SEQ + t) * DK + dkid] = f2bf(v);
            else
                Vt[((size_t)bh * DK + dkid) * SEQ + t] = f2bf(v);
        }
    }
}

// -------------------------------------------------- flash attention kernel
// grid = (T/128, B*H); 8 waves/block, each wave owns 16 query rows.
// K/V chunks (shared by all 8 waves) are async-staged to LDS, double-buffered.
__global__ __launch_bounds__(256) void attn_kernel(
    const __bf16* __restrict__ Q, const __bf16* __restrict__ K,
    const __bf16* __restrict__ Vt, const float* __restrict__ bias,
    __bf16* __restrict__ O)
{
    __shared__ __attribute__((aligned(16))) __bf16 Ks[2][32][64]; // 8 KB
    __shared__ __attribute__((aligned(16))) __bf16 Vs[2][64][32]; // 8 KB
    __shared__ __attribute__((aligned(16))) __bf16 Pb[8][16 * 32];// 8 KB

    const int tid  = threadIdx.x;
    const int w    = tid >> 5;
    const int lane = tid & 31;
    const int col  = lane & 15;
    const int hi   = lane >> 4;
    const int q0   = blockIdx.x * 128 + w * 16;
    const int bh   = blockIdx.y;
    const int b    = bh >> 4;
    const int h    = bh & 15;

    const __bf16* Qbh = Q  + (size_t)bh * SEQ * DK;
    const __bf16* Kbh = K  + (size_t)bh * SEQ * DK;
    const __bf16* Vbh = Vt + (size_t)bh * DK * SEQ;
    const float*  Bh  = bias + (size_t)h * SEQ * SEQ;   // broadcast over batch

    // 2 async b128 per thread per chunk: 32x64 K tile + 64x32 V^T tile
    auto stage = [&](int k0, int buf) {
        { const int r = tid >> 3, c = (tid & 7) * 8;
          async_copy16(Kbh + (size_t)(k0 + r) * DK + c, &Ks[buf][r][c]); }
        { const int r = tid >> 2, c = (tid & 3) * 8;
          async_copy16(Vbh + (size_t)r * SEQ + k0 + c, &Vs[buf][r][c]); }
    };

    // Q fragments (dk=64 -> two K=32 A-frags); Q already pre-scaled
    const v16bf qa0 = load_frag(Qbh + (size_t)q0 * DK,      DK);
    const v16bf qa1 = load_frag(Qbh + (size_t)q0 * DK + 32, DK);

    v8f acc[4] = {};
    float mrun[8], lrun[8];
#pragma unroll
    for (int j = 0; j < 8; ++j) { mrun[j] = -1e30f; lrun[j] = 0.0f; }

    __bf16* Pw = &Pb[w][0];

    stage(0, 0);
    for (int it = 0; it < SEQ / 32; ++it) {
        const int k0  = it * 32;
        const int buf = it & 1;
        if (it + 1 < SEQ / 32) { stage(k0 + 32, buf ^ 1); WAIT_ASYNC(2); }
        else                   { WAIT_ASYNC(0); }
        __syncthreads();

        // scores: two 16-key tiles, K=64 contraction each
        v16bf kb00 = load_frag(&Ks[buf][0][0],   64);
        v16bf kb01 = load_frag(&Ks[buf][0][32],  64);
        v16bf kb10 = load_frag(&Ks[buf][16][0],  64);
        v16bf kb11 = load_frag(&Ks[buf][16][32], 64);
        v8f s0 = {}; s0 = wmma_bf16(qa0, kb00, s0); s0 = wmma_bf16(qa1, kb01, s0);
        v8f s1 = {}; s1 = wmma_bf16(qa0, kb10, s1); s1 = wmma_bf16(qa1, kb11, s1);

        // fp32 bias + online softmax (row M=j+8*hi lives in one 16-lane half)
#pragma unroll
        for (int j = 0; j < 8; ++j) {
            const int qrow = q0 + j + hi * 8;
            const float* bp = Bh + (size_t)qrow * SEQ + k0 + col;
            float t0 = s0[j] + bp[0];
            float t1 = s1[j] + bp[16];

            float vmax = fmaxf(t0, t1);
            vmax = fmaxf(vmax, __shfl_xor(vmax, 1, 32));
            vmax = fmaxf(vmax, __shfl_xor(vmax, 2, 32));
            vmax = fmaxf(vmax, __shfl_xor(vmax, 4, 32));
            vmax = fmaxf(vmax, __shfl_xor(vmax, 8, 32));

            const float mn    = fmaxf(mrun[j], vmax);
            const float alpha = __expf(mrun[j] - mn);
            mrun[j] = mn;

            const float p0 = __expf(t0 - mn);
            const float p1 = __expf(t1 - mn);
            float rs = p0 + p1;
            rs += __shfl_xor(rs, 1, 32);
            rs += __shfl_xor(rs, 2, 32);
            rs += __shfl_xor(rs, 4, 32);
            rs += __shfl_xor(rs, 8, 32);
            lrun[j] = lrun[j] * alpha + rs;

            acc[0][j] *= alpha; acc[1][j] *= alpha;
            acc[2][j] *= alpha; acc[3][j] *= alpha;

            const int prow = (j + hi * 8) * 32;   // C-layout -> LDS P tile
            Pw[prow + col]      = f2bf(p0);
            Pw[prow + 16 + col] = f2bf(p1);
        }

        // P (re-read via A-layout swizzle) @ V
        v16bf pa = load_frag(Pw, 32);
#pragma unroll
        for (int n = 0; n < 4; ++n) {
            v16bf vb = load_frag(&Vs[buf][n * 16][0], 32);
            acc[n] = wmma_bf16(pa, vb, acc[n]);
        }
        __syncthreads();
    }

    // epilogue: normalize, store O as [B, T, H*dk] bf16
#pragma unroll
    for (int n = 0; n < 4; ++n) {
#pragma unroll
        for (int j = 0; j < 8; ++j) {
            const int t = q0 + j + hi * 8;
            const float v = acc[n][j] / lrun[j];
            O[((size_t)b * SEQ + t) * D_MODEL + h * DK + n * 16 + col] = f2bf(v);
        }
    }
}

// -------------------------------------------------- output projection GEMM
__global__ __launch_bounds__(256) void out_gemm(
    const __bf16* __restrict__ Obf, const __bf16* __restrict__ Wo,
    const float* __restrict__ bo, float* __restrict__ out)
{
    __shared__ __attribute__((aligned(16))) __bf16 As[2][128][32];
    __shared__ __attribute__((aligned(16))) __bf16 Bs[2][64][32];

    const int tid  = threadIdx.x;
    const int w    = tid >> 5;
    const int lane = tid & 31;
    const int col  = lane & 15;
    const int hi   = lane >> 4;
    const int rowB = blockIdx.x * 128;
    const int n0   = blockIdx.y * 64;

    auto stage = [&](int kc, int buf) {
#pragma unroll
        for (int s = 0; s < 2; ++s) {
            const int seg = tid * 2 + s;
            const int r = seg >> 2, c = (seg & 3) * 8;
            async_copy16(Obf + (size_t)(rowB + r) * D_MODEL + kc + c,
                         &As[buf][r][c]);
        }
        const int r = tid >> 2, c = (tid & 3) * 8;
        async_copy16(Wo + (size_t)(n0 + r) * D_MODEL + kc + c, &Bs[buf][r][c]);
    };

    v8f acc[4] = {};
    stage(0, 0);
    for (int it = 0; it < D_MODEL / 32; ++it) {
        const int buf = it & 1;
        if (it + 1 < D_MODEL / 32) { stage((it + 1) * 32, buf ^ 1); WAIT_ASYNC(3); }
        else                       { WAIT_ASYNC(0); }
        __syncthreads();

        v16bf a = load_frag(&As[buf][w * 16][0], 32);
#pragma unroll
        for (int n = 0; n < 4; ++n) {
            v16bf bb = load_frag(&Bs[buf][n * 16][0], 32);
            acc[n] = wmma_bf16(a, bb, acc[n]);
        }
        __syncthreads();
    }

    const int row0 = rowB + w * 16;
#pragma unroll
    for (int n = 0; n < 4; ++n) {
        const int   c  = n0 + n * 16 + col;
        const float bc = bo[c];
#pragma unroll
        for (int j = 0; j < 8; ++j) {
            const int row = row0 + j + hi * 8;
            out[(size_t)row * D_MODEL + c] = acc[n][j] + bc;
        }
    }
}

// --------------------------------------------------------------- launcher
extern "C" void kernel_launch(void* const* d_in, const int* in_sizes, int n_in,
                              void* d_out, int out_size, void* d_ws, size_t ws_size,
                              hipStream_t stream) {
    (void)in_sizes; (void)n_in; (void)out_size; (void)ws_size;

    const float* x  = (const float*)d_in[0];
    const float* ab = (const float*)d_in[1];
    const float* Wq = (const float*)d_in[2];
    const float* bq = (const float*)d_in[3];
    const float* Wk = (const float*)d_in[4];
    const float* bk = (const float*)d_in[5];
    const float* Wv = (const float*)d_in[6];
    const float* bv = (const float*)d_in[7];
    const float* Wo = (const float*)d_in[8];
    const float* bo = (const float*)d_in[9];
    float* out = (float*)d_out;

    char* ws = (char*)d_ws;
    size_t off = 0;
    auto take = [&](size_t bytes) { char* p = ws + off; off += bytes; return p; };

    __bf16* xbf = (__bf16*)take((size_t)NROWS * D_MODEL * 2);          // 8 MB
    __bf16* Wqb = (__bf16*)take((size_t)D_MODEL * D_MODEL * 2);        // 2 MB
    __bf16* Wkb = (__bf16*)take((size_t)D_MODEL * D_MODEL * 2);
    __bf16* Wvb = (__bf16*)take((size_t)D_MODEL * D_MODEL * 2);
    __bf16* Wob = (__bf16*)take((size_t)D_MODEL * D_MODEL * 2);
    __bf16* Qb  = (__bf16*)take((size_t)BATCH * NHEAD * SEQ * DK * 2); // 8 MB
    __bf16* Kb  = (__bf16*)take((size_t)BATCH * NHEAD * SEQ * DK * 2);
    __bf16* Vtb = (__bf16*)take((size_t)BATCH * NHEAD * DK * SEQ * 2);
    __bf16* Ob  = (__bf16*)take((size_t)NROWS * D_MODEL * 2);

    cvt_f32_to_bf16<<<1024, 256, 0, stream>>>(x,  xbf, NROWS * D_MODEL);
    cvt_f32_to_bf16<<<512,  256, 0, stream>>>(Wq, Wqb, D_MODEL * D_MODEL);
    cvt_f32_to_bf16<<<512,  256, 0, stream>>>(Wk, Wkb, D_MODEL * D_MODEL);
    cvt_f32_to_bf16<<<512,  256, 0, stream>>>(Wv, Wvb, D_MODEL * D_MODEL);
    cvt_f32_to_bf16<<<512,  256, 0, stream>>>(Wo, Wob, D_MODEL * D_MODEL);

    qkv_gemm<<<dim3(NROWS / 128, D_MODEL / 64, 3), 256, 0, stream>>>(
        xbf, Wqb, Wkb, Wvb, bq, bk, bv, Qb, Kb, Vtb);

    attn_kernel<<<dim3(SEQ / 128, BATCH * NHEAD), 256, 0, stream>>>(
        Qb, Kb, Vtb, ab, Ob);

    out_gemm<<<dim3(NROWS / 128, D_MODEL / 64), 256, 0, stream>>>(
        Ob, Wob, bo, out);
}
